// Interacted_User_Aggregation_12283606466573
// MI455X (gfx1250) — matrix operand order repaired and checked by reading, not compiled, not run on hardware
//
#include <hip/hip_runtime.h>

// MI455X / gfx1250 GraphSAGE-attention aggregation.
// Strategy: gather-bandwidth-bound problem (210MB random gather @ 23.3TB/s ~ 9us);
// do the two dense layers with f16 WMMA (f32 accum) so compute stays under the
// gather time. One 128-thread block (4 wave32) per node; all tensors staged in LDS.

typedef _Float16 v8h  __attribute__((ext_vector_type(8)));
typedef _Float16 v16h __attribute__((ext_vector_type(16)));
typedef float    v8f  __attribute__((ext_vector_type(8)));

#define KN   50     // neighbors per node
#define XSTR 136    // padded LDS row stride (halves) for 128-wide rows (272B = 17*16B -> bank rotation)
#define HSTR 72     // padded LDS row stride (halves) for 64-wide rows  (144B = 9*16B)

// --- prep: W1[128,64] -> W1^T f16 [64,128], W2[64,64] -> W2^T f16 [64,64] in d_ws ---
__global__ void prep_w_kernel(const float* __restrict__ W1, const float* __restrict__ W2,
                              _Float16* __restrict__ w1t, _Float16* __restrict__ w2t) {
  int idx = blockIdx.x * blockDim.x + threadIdx.x;
  int stride = gridDim.x * blockDim.x;
  for (int i = idx; i < 64 * 128; i += stride) {
    int n = i >> 7, k = i & 127;
    w1t[i] = (_Float16)W1[k * 64 + n];
  }
  for (int i = idx; i < 64 * 64; i += stride) {
    int n = i >> 6, k = i & 63;
    w2t[i] = (_Float16)W2[k * 64 + n];
  }
}

__global__ __launch_bounds__(128)
void gat_agg_kernel(const float* __restrict__ table,
                    const float* __restrict__ b1, const float* __restrict__ b2,
                    const float* __restrict__ W3, const float* __restrict__ b3,
                    const int* __restrict__ nodes, const int* __restrict__ nidx,
                    const unsigned char* __restrict__ nmask,
                    const _Float16* __restrict__ w1t, const _Float16* __restrict__ w2t,
                    float* __restrict__ out) {
  __shared__ __align__(16) _Float16 sW1T[64][XSTR];   // W1^T: row n, contiguous k=0..127
  __shared__ __align__(16) _Float16 sW2T[64][HSTR];   // W2^T: row n, contiguous k=0..63
  __shared__ __align__(16) _Float16 sX[64][XSTR];     // x = [E | U], rows 50..63 zero
  __shared__ __align__(16) _Float16 sH1[64][HSTR];    // relu(x@W1+b1) in f16
  __shared__ _Float16 sUh[64];
  __shared__ float sB1[64], sB2[64], sW3v[64], sS[64], sAtt[64];
  __shared__ int sNI[KN];

  const int tid  = threadIdx.x;
  const int node = blockIdx.x;

  // ---- phase A: scalars / indices / center embedding ----
  if (tid < 64) {
    int u = nodes[node];
    sUh[tid]  = (_Float16)table[(size_t)u * 64 + tid];
    sB1[tid]  = b1[tid];
    sB2[tid]  = b2[tid];
    sW3v[tid] = W3[tid];
  }
  if (tid < KN) sNI[tid] = nidx[node * KN + tid];
  __syncthreads();

  // ---- phase B: stage weights (from L2-resident ws) + gather neighbors ----
  {
    const v8h* s1 = (const v8h*)w1t;           // 1024 x 16B chunks
    for (int i = tid; i < 1024; i += 128) {
      int n = i >> 4, q = i & 15;
      *(v8h*)&sW1T[n][q * 8] = s1[i];
    }
    const v8h* s2 = (const v8h*)w2t;           // 512 x 16B chunks
    for (int i = tid; i < 512; i += 128) {
      int n = i >> 3, q = i & 7;
      *(v8h*)&sW2T[n][q * 8] = s2[i];
    }
    // gather E (random 256B rows from HBM), convert to f16 into x[:, 0:64]
    for (int i = tid; i < KN * 16; i += 128) {
      int k = i >> 4, c = i & 15;
      const float4 v = ((const float4*)(table + (size_t)sNI[k] * 64))[c];
      sX[k][c * 4 + 0] = (_Float16)v.x;
      sX[k][c * 4 + 1] = (_Float16)v.y;
      sX[k][c * 4 + 2] = (_Float16)v.z;
      sX[k][c * 4 + 3] = (_Float16)v.w;
    }
    // broadcast center U into x[:, 64:128]
    for (int i = tid; i < KN * 64; i += 128) {
      int k = i >> 6, j = i & 63;
      sX[k][64 + j] = sUh[j];
    }
    // zero padded rows 50..63 (tid spans all 128 columns)
    for (int r = KN; r < 64; ++r) sX[r][tid] = (_Float16)0.f;
  }
  __syncthreads();

  const int wv   = tid >> 5;     // wave id 0..3 -> owns rows 16*wv .. 16*wv+15
  const int lane = tid & 31;
  const int lh   = lane & 15;
  const int sel  = lane >> 4;    // which half-wave

  union AF { v16h v; v8h h[2]; };

  // ---- layer 1: h1 = relu(x @ W1 + b1), M=64 N=64 K=128 ----
  v8f c1[4] = {};
  const int arow = wv * 16 + lh;
  #pragma unroll
  for (int kt = 0; kt < 4; ++kt) {
    AF a;
    const int kb = kt * 32 + (sel ? 8 : 0);          // A frag: two 16B chunks @ kb, kb+16
    a.h[0] = *(const v8h*)&sX[arow][kb];
    a.h[1] = *(const v8h*)&sX[arow][kb + 16];
    const int kbb = kt * 32 + (sel ? 8 : 0);         // B frag: two 16B chunks @ kbb, kbb+16
    #pragma unroll
    for (int nt = 0; nt < 4; ++nt) {
      AF b;
      b.h[0] = *(const v8h*)&sW1T[nt * 16 + lh][kbb];
      b.h[1] = *(const v8h*)&sW1T[nt * 16 + lh][kbb + 16];
      c1[nt] = __builtin_amdgcn_wmma_f32_16x16x32_f16(
          false, a.v, false, b.v, (short)0, c1[nt], false, false);
    }
  }
  // bias + relu -> f16 h1 in LDS (each wave writes only its own rows)
  const int hrow = wv * 16 + sel * 8;
  #pragma unroll
  for (int nt = 0; nt < 4; ++nt) {
    const int col = nt * 16 + lh;
    const float bb = sB1[col];
    #pragma unroll
    for (int r = 0; r < 8; ++r) {
      float hv = fmaxf(c1[nt][r] + bb, 0.f);
      sH1[hrow + r][col] = (_Float16)hv;
    }
  }

  // ---- layer 2: h2 = relu(h1 @ W2 + b2), M=64 N=64 K=64 (A rows are this wave's own) ----
  v8f c2[4] = {};
  #pragma unroll
  for (int kt = 0; kt < 2; ++kt) {
    AF a;
    const int kb = kt * 32 + (sel ? 8 : 0);
    a.h[0] = *(const v8h*)&sH1[arow][kb];
    a.h[1] = *(const v8h*)&sH1[arow][kb + 16];
    const int kbb = kt * 32 + (sel ? 8 : 0);
    #pragma unroll
    for (int nt = 0; nt < 4; ++nt) {
      AF b;
      b.h[0] = *(const v8h*)&sW2T[nt * 16 + lh][kbb];
      b.h[1] = *(const v8h*)&sW2T[nt * 16 + lh][kbb + 16];
      c2[nt] = __builtin_amdgcn_wmma_f32_16x16x32_f16(
          false, a.v, false, b.v, (short)0, c2[nt], false, false);
    }
  }

  // ---- s = relu(h2) @ W3 + b3, straight from C fragments ----
  const float b3v = b3[0];
  #pragma unroll
  for (int r = 0; r < 8; ++r) {
    float p = 0.f;
    #pragma unroll
    for (int nt = 0; nt < 4; ++nt) {
      const int col = nt * 16 + lh;
      float hv = fmaxf(c2[nt][r] + sB2[col], 0.f);
      p += hv * sW3v[col];
    }
    // reduce across the 16-lane half (xor <= 8 stays within the half)
    p += __shfl_xor(p, 8);
    p += __shfl_xor(p, 4);
    p += __shfl_xor(p, 2);
    p += __shfl_xor(p, 1);
    if (lh == 0) sS[hrow + r] = p + b3v;
  }
  __syncthreads();

  // ---- masked softmax over K (wave 0) ----
  if (wv == 0) {
    const unsigned char* m = nmask + (size_t)node * KN;
    const int k0 = lane, k1 = lane + 32;
    float v0 = (k0 < KN && m[k0]) ? sS[k0] : -1e30f;
    float v1 = (k1 < KN && m[k1]) ? sS[k1] : -1e30f;
    float mx = fmaxf(v0, v1);
    #pragma unroll
    for (int off = 16; off >= 1; off >>= 1) mx = fmaxf(mx, __shfl_xor(mx, off));
    float e0 = (v0 > -1e29f) ? __expf(v0 - mx) : 0.f;
    float e1 = (v1 > -1e29f) ? __expf(v1 - mx) : 0.f;
    float se = e0 + e1;
    #pragma unroll
    for (int off = 16; off >= 1; off >>= 1) se += __shfl_xor(se, off);
    const float inv = 1.f / se;
    sAtt[k0] = e0 * inv;
    sAtt[k1] = e1 * inv;
  }
  __syncthreads();

  // ---- out[n] = att @ E ----
  if (tid < 64) {
    float acc = 0.f;
    #pragma unroll 5
    for (int k = 0; k < KN; ++k) acc += sAtt[k] * (float)sX[k][tid];
    out[(size_t)node * 64 + tid] = acc;
  }
}

extern "C" void kernel_launch(void* const* d_in, const int* in_sizes, int n_in,
                              void* d_out, int out_size, void* d_ws, size_t ws_size,
                              hipStream_t stream) {
  const float* table = (const float*)d_in[0];
  const float* W1    = (const float*)d_in[1];
  const float* b1    = (const float*)d_in[2];
  const float* W2    = (const float*)d_in[3];
  const float* b2    = (const float*)d_in[4];
  const float* W3    = (const float*)d_in[5];
  const float* b3    = (const float*)d_in[6];
  const int*   nodes = (const int*)d_in[7];
  const int*   nidx  = (const int*)d_in[8];
  const unsigned char* nmask = (const unsigned char*)d_in[9];
  const int nNodes = in_sizes[7];

  // workspace: 24KB of transposed f16 weights
  _Float16* w1t = (_Float16*)d_ws;
  _Float16* w2t = w1t + 64 * 128;

  prep_w_kernel<<<16, 256, 0, stream>>>(W1, W2, w1t, w2t);
  gat_agg_kernel<<<nNodes, 128, 0, stream>>>(table, b1, b2, W3, b3,
                                             nodes, nidx, nmask, w1t, w2t,
                                             (float*)d_out);
}
